// MAEGINDecoder_17162689315601
// MI455X (gfx1250) — matrix operand order repaired and checked by reading, not compiled, not run on hardware
//
#include <hip/hip_runtime.h>
#include <hip/hip_bf16.h>

// ---------------- problem constants (from reference) ----------------
constexpr int N_NODES = 20000;
constexpr int N_EDGES = 640000;
constexpr int HIDDEN  = 512;
constexpr int VOCAB   = 8192;
constexpr int MIDDLE  = HIDDEN + (VOCAB - HIDDEN) / 2;  // 4352

// ---------------- vector types ----------------
typedef __attribute__((ext_vector_type(16))) __bf16 v16bf;
typedef __attribute__((ext_vector_type(8)))  __bf16 v8bf;
typedef __attribute__((ext_vector_type(4)))  __bf16 v4bf;
typedef __attribute__((ext_vector_type(8)))  float  v8f;

// float -> bf16, round-to-nearest-even
__device__ __forceinline__ __bf16 f2bf(float f) {
    unsigned u = __builtin_bit_cast(unsigned, f);
    unsigned r = u + 0x7FFFu + ((u >> 16) & 1u);
    unsigned short h = (unsigned short)(r >> 16);
    return __builtin_bit_cast(__bf16, h);
}

__device__ __forceinline__ v16bf join16(v8bf lo, v8bf hi) {
    v16bf r;
#pragma unroll
    for (int i = 0; i < 8; ++i) { r[i] = lo[i]; r[8 + i] = hi[i]; }
    return r;
}

// ---------------- kernel 1: h = x (f32 copy, float4) ----------------
__global__ __launch_bounds__(256) void copy_f32(const float* __restrict__ src,
                                                float* __restrict__ dst,
                                                size_t n4) {
    size_t i = (size_t)blockIdx.x * blockDim.x + threadIdx.x;
    size_t stride = (size_t)gridDim.x * blockDim.x;
    const float4* s = (const float4*)src;
    float4* d = (float4*)dst;
    for (; i < n4; i += stride) d[i] = s[i];
}

// ---------------- kernel 2: scatter-add over edges ----------------
// one wave (32 lanes) per edge; h[dst] += x[src]
__global__ __launch_bounds__(256) void gin_scatter(const float* __restrict__ x,
                                                   const long long* __restrict__ ei,
                                                   float* __restrict__ h) {
    int lane = threadIdx.x & 31;
    int waveInBlock = threadIdx.x >> 5;
    long long e = (long long)blockIdx.x * (blockDim.x >> 5) + waveInBlock;
    if (e >= N_EDGES) return;
    long long s = ei[e];
    long long d = ei[(long long)N_EDGES + e];
    const float* xs = x + (size_t)s * HIDDEN;
    float* hd = h + (size_t)d * HIDDEN;
#pragma unroll
    for (int c = 0; c < HIDDEN / 32; ++c)
        atomicAdd(&hd[lane + 32 * c], xs[lane + 32 * c]);
}

// ---------------- kernel 3: f32 -> bf16 convert (4-wide) ----------------
__global__ __launch_bounds__(256) void cvt_f32_bf16(const float* __restrict__ src,
                                                    __bf16* __restrict__ dst,
                                                    size_t n) {
    size_t i = ((size_t)blockIdx.x * blockDim.x + threadIdx.x) * 4;
    size_t stride = (size_t)gridDim.x * blockDim.x * 4;
    for (; i < n; i += stride) {
        float4 v = *(const float4*)(src + i);
        v4bf o;
        o[0] = f2bf(v.x); o[1] = f2bf(v.y); o[2] = f2bf(v.z); o[3] = f2bf(v.w);
        *(v4bf*)(dst + i) = o;
    }
}

// ---------------- WMMA GEMM with async-LDS double buffering ----------------
// D[M,N] = A[M,K] * W[N,K]^T + bias[N]
// block = 256 threads = 8 waves (4 in M x 2 in N); wave tile = 32x64 (8 accums)
// block tile = 128(M) x 128(N), K staged in chunks of 32 via
// GLOBAL_LOAD_ASYNC_TO_LDS_B128 (ASYNCcnt) into double-buffered padded LDS.
template <bool OUT_BF16>
__global__ __launch_bounds__(256) void gemm_bf16_wmma(const __bf16* __restrict__ A,
                                                      const __bf16* __restrict__ W,
                                                      const float* __restrict__ bias,
                                                      void* __restrict__ Out,
                                                      int M, int N, int K) {
    constexpr int BM = 128, BN = 128, BK = 32;
    constexpr int LDP = 40;  // padded LDS row stride in elements (80B) -> conflict-free b128 reads

    __shared__ __bf16 Abuf[2][BM * LDP];
    __shared__ __bf16 Bbuf[2][BN * LDP];

    const int tid  = threadIdx.x;
    const int lane = tid & 31;
    const int wave = tid >> 5;
    const int l15  = lane & 15;
    const int kgrp = lane >> 4;  // 0 or 1

    const int blockM = blockIdx.x * BM;
    const int blockN = blockIdx.y * BN;
    const int mWave  = (wave & 3) * 32;   // 4 waves along M
    const int nWave  = (wave >> 2) * 64;  // 2 waves along N

    // LDS byte offsets (addrspace(3) offsets live in the low 32 bits of the flat addr)
    const unsigned ldsA = (unsigned)(unsigned long long)(&Abuf[0][0]);
    const unsigned ldsB = (unsigned)(unsigned long long)(&Bbuf[0][0]);

    // copy slots: 128 rows x 4 chunks(16B) per tile = 512 chunks; 2 per thread per tile
    int rowArr[2], chkArr[2];
#pragma unroll
    for (int i = 0; i < 2; ++i) {
        int id = tid + 256 * i;
        rowArr[i] = id >> 2;
        chkArr[i] = id & 3;
    }

    auto issue_stage = [&](int k0, int buf) {
#pragma unroll
        for (int i = 0; i < 2; ++i) {  // A tile
            int r = rowArr[i], c = chkArr[i];
            int rg = blockM + r; if (rg >= M) rg = M - 1;
            unsigned goff = (unsigned)(((size_t)rg * K + (size_t)(k0 + c * 8)) * 2u);
            unsigned loff = ldsA + (unsigned)((buf * BM * LDP + r * LDP + c * 8) * 2);
            asm volatile("global_load_async_to_lds_b128 %0, %1, %2"
                         :: "v"(loff), "v"(goff), "s"(A) : "memory");
        }
#pragma unroll
        for (int i = 0; i < 2; ++i) {  // B tile (N is a multiple of 128 -> no clamp)
            int r = rowArr[i], c = chkArr[i];
            int rg = blockN + r;
            unsigned goff = (unsigned)(((size_t)rg * K + (size_t)(k0 + c * 8)) * 2u);
            unsigned loff = ldsB + (unsigned)((buf * BN * LDP + r * LDP + c * 8) * 2);
            asm volatile("global_load_async_to_lds_b128 %0, %1, %2"
                         :: "v"(loff), "v"(goff), "s"(W) : "memory");
        }
    };

    v8f acc[2][4];
#pragma unroll
    for (int i = 0; i < 2; ++i)
#pragma unroll
        for (int j = 0; j < 4; ++j) acc[i][j] = (v8f){};

    const int nsteps = K / BK;
    issue_stage(0, 0);

    for (int s = 0; s < nsteps; ++s) {
        const int buf = s & 1;
        if (s + 1 < nsteps) {
            issue_stage((s + 1) * BK, buf ^ 1);
            // async loads complete in order: <=4 outstanding => stage s resident
            asm volatile("s_wait_asynccnt 4" ::: "memory");
        } else {
            asm volatile("s_wait_asynccnt 0" ::: "memory");
        }
        __syncthreads();  // all waves' stage-s data visible

        const __bf16* Ab = &Abuf[buf][0];
        const __bf16* Bb = &Bbuf[buf][0];

        v16bf fa[2], fb[4];
#pragma unroll
        for (int i = 0; i < 2; ++i) {
            const __bf16* p = Ab + (mWave + 16 * i + l15) * LDP + kgrp * 8;
            fa[i] = join16(*(const v8bf*)p, *(const v8bf*)(p + 16));
        }
#pragma unroll
        for (int j = 0; j < 4; ++j) {
            const __bf16* p = Bb + (nWave + 16 * j + l15) * LDP + kgrp * 8;
            fb[j] = join16(*(const v8bf*)p, *(const v8bf*)(p + 16));
        }
#pragma unroll
        for (int i = 0; i < 2; ++i)
#pragma unroll
            for (int j = 0; j < 4; ++j)
                acc[i][j] = __builtin_amdgcn_wmma_f32_16x16x32_bf16(
                    false, fa[i], false, fb[j], (short)0, acc[i][j], false, false);

        __syncthreads();  // protect buf before it is refilled next iteration
    }

    // epilogue: C/D layout -> VGPR r holds row (r + 8*kgrp), col = l15 within tile
    float*  outF = (float*)Out;
    __bf16* outB = (__bf16*)Out;
#pragma unroll
    for (int i = 0; i < 2; ++i) {
#pragma unroll
        for (int j = 0; j < 4; ++j) {
            const int n  = blockN + nWave + 16 * j + l15;  // always < N (N % 128 == 0)
            const float bv = bias[n];
#pragma unroll
            for (int r = 0; r < 8; ++r) {
                const int m = blockM + mWave + 16 * i + r + 8 * kgrp;
                if (m < M) {
                    float v = acc[i][j][r] + bv;
                    if (OUT_BF16) outB[(size_t)m * N + n] = f2bf(v);
                    else          outF[(size_t)m * N + n] = v;
                }
            }
        }
    }
}

// ---------------- launcher ----------------
extern "C" void kernel_launch(void* const* d_in, const int* in_sizes, int n_in,
                              void* d_out, int out_size, void* d_ws, size_t ws_size,
                              hipStream_t stream) {
    const float* x      = (const float*)d_in[0];
    const long long* ei = (const long long*)d_in[1];
    const float* W1     = (const float*)d_in[2];
    const float* b1     = (const float*)d_in[3];
    const float* W2     = (const float*)d_in[4];
    const float* b2     = (const float*)d_in[5];
    float* out          = (float*)d_out;

    // workspace layout (256B aligned slabs)
    char* base = (char*)d_ws;
    size_t off = 0;
    auto alloc = [&](size_t bytes) {
        size_t o = off;
        off = (off + bytes + 255) & ~(size_t)255;
        return o;
    };
    float*  h_f32  = (float*) (base + alloc((size_t)N_NODES * HIDDEN * 4));
    __bf16* h_bf   = (__bf16*)(base + alloc((size_t)N_NODES * HIDDEN * 2));
    __bf16* w1_bf  = (__bf16*)(base + alloc((size_t)MIDDLE * HIDDEN * 2));
    __bf16* w2_bf  = (__bf16*)(base + alloc((size_t)VOCAB * MIDDLE * 2));
    __bf16* mid_bf = (__bf16*)(base + alloc((size_t)N_NODES * MIDDLE * 2));
    (void)ws_size;

    // 1) h = x
    copy_f32<<<2048, 256, 0, stream>>>(x, h_f32, (size_t)N_NODES * HIDDEN / 4);

    // weight converts (independent of scatter)
    cvt_f32_bf16<<<2048, 256, 0, stream>>>(W1, w1_bf, (size_t)MIDDLE * HIDDEN);
    cvt_f32_bf16<<<4096, 256, 0, stream>>>(W2, w2_bf, (size_t)VOCAB * MIDDLE);

    // 2) h += segment_sum(x[src] -> dst)   (one wave per edge)
    gin_scatter<<<N_EDGES / 8, 256, 0, stream>>>(x, ei, h_f32);

    // 3) h -> bf16
    cvt_f32_bf16<<<4096, 256, 0, stream>>>(h_f32, h_bf, (size_t)N_NODES * HIDDEN);

    // 4) mid = h @ W1^T + b1 -> bf16   [20000 x 4352]
    {
        dim3 grid((N_NODES + 127) / 128, MIDDLE / 128);
        gemm_bf16_wmma<true><<<grid, 256, 0, stream>>>(h_bf, w1_bf, b1, mid_bf,
                                                       N_NODES, MIDDLE, HIDDEN);
    }

    // 5) out = mid @ W2^T + b2 -> f32  [20000 x 8192]
    {
        dim3 grid((N_NODES + 127) / 128, VOCAB / 128);
        gemm_bf16_wmma<false><<<grid, 256, 0, stream>>>(mid_bf, w2_bf, b2, out,
                                                        N_NODES, VOCAB, MIDDLE);
    }
}
